// Pilgrim_76888504533359
// MI455X (gfx1250) — matrix operand order repaired and checked by reading, not compiled, not run on hardware
//
#include <hip/hip_runtime.h>

typedef __attribute__((ext_vector_type(16))) __bf16 v16bf;
typedef __attribute__((ext_vector_type(8)))  __bf16 v8bf;
typedef __attribute__((ext_vector_type(8)))  float  v8f;

#define BN_EPS 1e-5f

// ---- CDNA5 async global->LDS copy (16B per lane), tracked by ASYNCcnt ----
__device__ __forceinline__ void async_copy16(const __bf16* g, __bf16* l) {
  unsigned laddr = (unsigned)(unsigned long long)l;   // LDS byte offset (low 32 bits of aperture addr)
  asm volatile("global_load_async_to_lds_b128 %0, %1, off"
               :: "v"(laddr), "v"(g) : "memory");
}
#define ASYNC_WAIT(n) asm volatile("s_wait_asynccnt %0" :: "n"(n) : "memory")

// ---------------------------------------------------------------------------
// w1 [R=4096, C=2304] f32  ->  w1t [2304, 4096] bf16   (tiled transpose)
// ---------------------------------------------------------------------------
__global__ __launch_bounds__(256)
void k_transpose_bf16(const float* __restrict__ in, __bf16* __restrict__ out,
                      int R, int Ccols) {
  __shared__ float tile[32][33];
  int c0 = blockIdx.x * 32;
  int r0 = blockIdx.y * 32;
  int tx = threadIdx.x, ty = threadIdx.y;          // block (32, 8)
#pragma unroll
  for (int i = 0; i < 32; i += 8)
    tile[ty + i][tx] = in[(size_t)(r0 + ty + i) * Ccols + (c0 + tx)];
  __syncthreads();
#pragma unroll
  for (int i = 0; i < 32; i += 8)
    out[(size_t)(c0 + ty + i) * R + (r0 + tx)] = (__bf16)tile[tx][ty + i];
}

// ---------------------------------------------------------------------------
// f32 -> bf16 elementwise convert (weights)
// ---------------------------------------------------------------------------
__global__ void k_f32_to_bf16(const float* __restrict__ in,
                              __bf16* __restrict__ out, size_t n) {
  size_t i = (size_t)blockIdx.x * blockDim.x + threadIdx.x;
  if (i < n) out[i] = (__bf16)in[i];
}

// ---------------------------------------------------------------------------
// Fold eval-mode BN (+ preceding linear bias) into y = s*x + t
// ---------------------------------------------------------------------------
__global__ void k_bnprep(const float* __restrict__ g, const float* __restrict__ be,
                         const float* __restrict__ m, const float* __restrict__ v,
                         const float* __restrict__ bias,
                         float* __restrict__ s, float* __restrict__ t, int n) {
  int i = blockIdx.x * blockDim.x + threadIdx.x;
  if (i < n) {
    float sc = g[i] * rsqrtf(v[i] + BN_EPS);
    s[i] = sc;
    t[i] = be[i] - sc * m[i] + sc * bias[i];
  }
}

// ---------------------------------------------------------------------------
// Embedding-bag layer 1: x1[b,j] = relu(s1[j] * sum_n w1t[n*48+z[b,n], j] + t1[j])
// ---------------------------------------------------------------------------
__global__ __launch_bounds__(256)
void k_embed(const int* __restrict__ z, const __bf16* __restrict__ w1t,
             const float* __restrict__ s, const float* __restrict__ t,
             __bf16* __restrict__ x1) {
  __shared__ int zs[48];
  const int b = blockIdx.x;
  const int tid = threadIdx.x;
  if (tid < 48) zs[tid] = z[b * 48 + tid];
  __syncthreads();

  const int j0 = tid * 16;
  float acc[16];
#pragma unroll
  for (int c = 0; c < 16; ++c) acc[c] = 0.f;

  for (int n = 0; n < 48; ++n) {
    const v8bf* p = (const v8bf*)(w1t + (size_t)(n * 48 + zs[n]) * 4096 + j0);
    v8bf a0 = p[0], a1 = p[1];
#pragma unroll
    for (int c = 0; c < 8; ++c) {
      acc[c]     += (float)a0[c];
      acc[c + 8] += (float)a1[c];
    }
  }

  v8bf o0, o1;
#pragma unroll
  for (int c = 0; c < 16; ++c) {
    float y = s[j0 + c] * acc[c] + t[j0 + c];
    y = y > 0.f ? y : 0.f;
    if (c < 8) o0[c] = (__bf16)y; else o1[c - 8] = (__bf16)y;
  }
  v8bf* op = (v8bf*)(x1 + (size_t)b * 4096 + j0);
  op[0] = o0;
  op[1] = o1;
}

// ---------------------------------------------------------------------------
// bf16 GEMM, fused BN/ReLU/residual epilogue:
//   out[b,o] = relu( s[o] * (A[b,:] . W[o,:]) + t[o] (+ R[b,o]) )
// Block tile 128x128x64, double-buffered LDS fed by async global->LDS copies.
// 8 waves; wave tile 32x64 = 2x4 WMMA accumulators; 16 WMMAs / wave / K-iter.
// ---------------------------------------------------------------------------
template<int K>
__global__ __launch_bounds__(256)
void k_gemm(const __bf16* __restrict__ A, const __bf16* __restrict__ W,
            const float* __restrict__ s, const float* __restrict__ t,
            const __bf16* __restrict__ R, __bf16* __restrict__ out, int O) {
  __shared__ alignas(32) __bf16 As[2][128 * 64];   // 16KB x2
  __shared__ alignas(32) __bf16 Bs[2][128 * 64];   // 16KB x2

  const int tid  = threadIdx.x;
  const int lane = tid & 31;
  const int wave = tid >> 5;       // 0..7
  const int wm   = wave & 3;       // M offset 32*wm
  const int wn   = wave >> 2;      // N offset 64*wn
  const int bm   = blockIdx.y * 128;
  const int bn   = blockIdx.x * 128;

  const int lrow = lane & 15;
  const int hi   = lane >> 4;      // half-wave select

  v8f acc[2][4];
#pragma unroll
  for (int i = 0; i < 2; ++i)
#pragma unroll
    for (int j = 0; j < 4; ++j)
#pragma unroll
      for (int e = 0; e < 8; ++e) acc[i][j][e] = 0.f;

  // issue 8 async B128 copies for one 128x64 A-tile + 128x64 B-tile
  auto load_tile = [&](int buf, int k0) {
#pragma unroll
    for (int u = 0; u < 4; ++u) {
      int unit = tid + u * 256;           // 0..1023
      int row  = unit >> 3;               // 0..127
      int c8   = (unit & 7) * 8;          // 0..56
      async_copy16(&A[(size_t)(bm + row) * K + k0 + c8], &As[buf][row * 64 + c8]);
      async_copy16(&W[(size_t)(bn + row) * K + k0 + c8], &Bs[buf][row * 64 + c8]);
    }
  };

  constexpr int NITER = K / 64;
  load_tile(0, 0);

  for (int it = 0; it < NITER; ++it) {
    if (it + 1 < NITER) {
      load_tile((it + 1) & 1, (it + 1) * 64);
      ASYNC_WAIT(8);                 // previous tile's 8 copies retired, next 8 in flight
    } else {
      ASYNC_WAIT(0);
    }
    __syncthreads();                 // all waves' copies for current buffer visible

    const __bf16* Ab = As[it & 1];
    const __bf16* Bb = Bs[it & 1];
#pragma unroll
    for (int ks = 0; ks < 2; ++ks) {
      // A fragments: 16x32 bf16, row = lane%16, k = 16*hi + s (contiguous, same as B)
      v16bf af[2];
#pragma unroll
      for (int mt = 0; mt < 2; ++mt) {
        const __bf16* base = &Ab[(wm * 32 + mt * 16 + lrow) * 64 + ks * 32 + hi * 16];
        union { v16bf v; v8bf h[2]; } u;
        u.h[0] = *(const v8bf*)(base);
        u.h[1] = *(const v8bf*)(base + 8);
        af[mt] = u.v;
      }
      // B fragments: 32x16 bf16 (lane = column, K contiguous)
      v16bf bfrag[4];
#pragma unroll
      for (int nt = 0; nt < 4; ++nt) {
        const __bf16* base = &Bb[(wn * 64 + nt * 16 + lrow) * 64 + ks * 32 + hi * 16];
        bfrag[nt] = *(const v16bf*)base;
      }
#pragma unroll
      for (int mt = 0; mt < 2; ++mt)
#pragma unroll
        for (int nt = 0; nt < 4; ++nt)
          acc[mt][nt] = __builtin_amdgcn_wmma_f32_16x16x32_bf16(
              false, af[mt], false, bfrag[nt], (short)0, acc[mt][nt], false, false);
    }
    if (it + 1 < NITER) __syncthreads();   // LDS reads done before buffer reuse
  }

  // ---- epilogue: BN scale/shift, optional residual, ReLU, bf16 store ----
#pragma unroll
  for (int nt = 0; nt < 4; ++nt) {
    int col = bn + wn * 64 + nt * 16 + lrow;
    float sc = s[col], tc = t[col];
#pragma unroll
    for (int mt = 0; mt < 2; ++mt) {
      int rbase = bm + wm * 32 + mt * 16 + hi * 8;
#pragma unroll
      for (int r = 0; r < 8; ++r) {
        int row = rbase + r;
        float y = sc * acc[mt][nt][r] + tc;
        if (R) y += (float)R[(size_t)row * O + col];
        y = y > 0.f ? y : 0.f;
        out[(size_t)row * O + col] = (__bf16)y;
      }
    }
  }
}

// ---------------------------------------------------------------------------
// Output layer: out[b] = X[b,:] . wout + bout   (f32 result)
// ---------------------------------------------------------------------------
__global__ __launch_bounds__(256)
void k_dot(const __bf16* __restrict__ X, const float* __restrict__ wout,
           const float* __restrict__ bout, float* __restrict__ out) {
  __shared__ float red[256];
  const int b = blockIdx.x;
  const int tid = threadIdx.x;
  float acc = 0.f;
  for (int j = tid; j < 2048; j += 256)
    acc += (float)X[(size_t)b * 2048 + j] * wout[j];
  red[tid] = acc;
  __syncthreads();
  for (int sft = 128; sft > 0; sft >>= 1) {
    if (tid < sft) red[tid] += red[tid + sft];
    __syncthreads();
  }
  if (tid == 0) out[b] = red[0] + bout[0];
}

// ---------------------------------------------------------------------------
extern "C" void kernel_launch(void* const* d_in, const int* in_sizes, int n_in,
                              void* d_out, int out_size, void* d_ws, size_t ws_size,
                              hipStream_t stream) {
  (void)in_sizes; (void)n_in; (void)out_size; (void)ws_size;

  const int*   z    = (const int*)  d_in[0];
  const float* w1   = (const float*)d_in[1];
  const float* b1   = (const float*)d_in[2];
  const float* g1   = (const float*)d_in[3];
  const float* be1  = (const float*)d_in[4];
  const float* m1   = (const float*)d_in[5];
  const float* v1   = (const float*)d_in[6];
  const float* w2   = (const float*)d_in[7];
  const float* b2   = (const float*)d_in[8];
  const float* g2   = (const float*)d_in[9];
  const float* be2  = (const float*)d_in[10];
  const float* m2   = (const float*)d_in[11];
  const float* v2   = (const float*)d_in[12];
  const float* rw1  = (const float*)d_in[13];
  const float* rb1  = (const float*)d_in[14];
  const float* rg1  = (const float*)d_in[15];
  const float* rbe1 = (const float*)d_in[16];
  const float* rm1  = (const float*)d_in[17];
  const float* rv1  = (const float*)d_in[18];
  const float* rw2  = (const float*)d_in[19];
  const float* rb2  = (const float*)d_in[20];
  const float* rg2  = (const float*)d_in[21];
  const float* rbe2 = (const float*)d_in[22];
  const float* rm2  = (const float*)d_in[23];
  const float* rv2  = (const float*)d_in[24];
  const float* wout = (const float*)d_in[25];
  const float* bout = (const float*)d_in[26];

  const int B = 32768, HD1 = 4096, HD2 = 2048, NC = 48 * 48;

  char* ws = (char*)d_ws;
  size_t off = 0;
  auto take = [&](size_t bytes) -> void* {
    void* p = (void*)(ws + off);
    off += (bytes + 255) & ~(size_t)255;
    return p;
  };

  __bf16* w1t  = (__bf16*)take((size_t)NC * HD1 * 2);
  __bf16* w2b  = (__bf16*)take((size_t)HD2 * HD1 * 2);
  __bf16* rw1b = (__bf16*)take(2ull * HD2 * HD2 * 2);
  __bf16* rw2b = (__bf16*)take(2ull * HD2 * HD2 * 2);
  float* s1  = (float*)take(HD1 * 4);      float* t1  = (float*)take(HD1 * 4);
  float* s2  = (float*)take(HD2 * 4);      float* t2  = (float*)take(HD2 * 4);
  float* rs1 = (float*)take(2 * HD2 * 4);  float* rt1 = (float*)take(2 * HD2 * 4);
  float* rs2 = (float*)take(2 * HD2 * 4);  float* rt2 = (float*)take(2 * HD2 * 4);
  __bf16* X1 = (__bf16*)take((size_t)B * HD1 * 2);
  __bf16* Xa = (__bf16*)take((size_t)B * HD2 * 2);
  __bf16* Xb = (__bf16*)take((size_t)B * HD2 * 2);
  __bf16* H  = (__bf16*)take((size_t)B * HD2 * 2);

  // ---- weight prep (transpose + bf16 convert, BN folding) ----
  k_transpose_bf16<<<dim3(NC / 32, HD1 / 32), dim3(32, 8), 0, stream>>>(w1, w1t, HD1, NC);
  {
    size_t n = (size_t)HD2 * HD1;
    k_f32_to_bf16<<<(unsigned)((n + 255) / 256), 256, 0, stream>>>(w2, w2b, n);
    n = 2ull * HD2 * HD2;
    k_f32_to_bf16<<<(unsigned)((n + 255) / 256), 256, 0, stream>>>(rw1, rw1b, n);
    k_f32_to_bf16<<<(unsigned)((n + 255) / 256), 256, 0, stream>>>(rw2, rw2b, n);
  }
  k_bnprep<<<HD1 / 256, 256, 0, stream>>>(g1, be1, m1, v1, b1, s1, t1, HD1);
  k_bnprep<<<HD2 / 256, 256, 0, stream>>>(g2, be2, m2, v2, b2, s2, t2, HD2);
  for (int i = 0; i < 2; ++i) {
    k_bnprep<<<HD2 / 256, 256, 0, stream>>>(rg1 + i * HD2, rbe1 + i * HD2, rm1 + i * HD2,
                                            rv1 + i * HD2, rb1 + i * HD2,
                                            rs1 + i * HD2, rt1 + i * HD2, HD2);
    k_bnprep<<<HD2 / 256, 256, 0, stream>>>(rg2 + i * HD2, rbe2 + i * HD2, rm2 + i * HD2,
                                            rv2 + i * HD2, rb2 + i * HD2,
                                            rs2 + i * HD2, rt2 + i * HD2, HD2);
  }

  // ---- layer 1: embedding-bag + BN + ReLU -> X1 (bf16) ----
  k_embed<<<B, 256, 0, stream>>>(z, w1t, s1, t1, X1);

  // ---- layer 2: X1 @ w2.T -> Xa ----
  dim3 g(HD2 / 128, B / 128);
  k_gemm<4096><<<g, 256, 0, stream>>>(X1, w2b, s2, t2, nullptr, Xa, HD2);

  // ---- residual block 0: Xa -> H -> Xb (+Xa) ----
  k_gemm<2048><<<g, 256, 0, stream>>>(Xa, rw1b, rs1, rt1, nullptr, H, HD2);
  k_gemm<2048><<<g, 256, 0, stream>>>(H, rw2b, rs2, rt2, Xa, Xb, HD2);

  // ---- residual block 1: Xb -> H -> Xa (+Xb) ----
  k_gemm<2048><<<g, 256, 0, stream>>>(Xb, rw1b + (size_t)HD2 * HD2, rs1 + HD2, rt1 + HD2,
                                      nullptr, H, HD2);
  k_gemm<2048><<<g, 256, 0, stream>>>(H, rw2b + (size_t)HD2 * HD2, rs2 + HD2, rt2 + HD2,
                                      Xb, Xa, HD2);

  // ---- output head ----
  k_dot<<<B, 256, 0, stream>>>(Xa, wout, bout, (float*)d_out);
}
